// MultiHeadAttention_36876589204288
// MI455X (gfx1250) — matrix operand order repaired and checked by reading, not compiled, not run on hardware
//
#include <hip/hip_runtime.h>
#include <hip/hip_bf16.h>

#define D_MODEL 1024
#define NHEAD   16
#define DQKV    64
#define SEQ     2048
#define BATCH   2
#define MTOT    (BATCH*SEQ)   // 4096 rows total
#define LN_EPS  1e-5f

typedef __bf16 bf16_t;
typedef bf16_t v16bf __attribute__((ext_vector_type(16)));
typedef bf16_t v8bf  __attribute__((ext_vector_type(8)));
typedef float  v8f   __attribute__((ext_vector_type(8)));
typedef unsigned int u32x4 __attribute__((ext_vector_type(4)));
typedef int i32x4 __attribute__((ext_vector_type(4)));
typedef int i32x8 __attribute__((ext_vector_type(8)));

__device__ __forceinline__ v8f vzero8() {
    v8f z;
#pragma unroll
    for (int i = 0; i < 8; ++i) z[i] = 0.0f;
    return z;
}

__device__ __forceinline__ v8bf ld8(const bf16_t* p) { return *(const v8bf*)p; }

__device__ __forceinline__ v16bf cat16(v8bf lo, v8bf hi) {
    v16bf r;
#pragma unroll
    for (int i = 0; i < 8; ++i) { r[i] = lo[i]; r[i + 8] = hi[i]; }
    return r;
}

__device__ __forceinline__ v8f wmma_bf16(v16bf a, v16bf b, v8f c) {
    return __builtin_amdgcn_wmma_f32_16x16x32_bf16(
        false, a, false, b, (short)0, c, false, false);
}

// ---------------------------------------------------------------- TDM helper
// 2D tile DMA global->LDS via Tensor Data Mover (D# per CDNA5 ISA).
// tile0 = row length (elements), tile1 = #rows (0 => 1D), stride0 in elements.
// Elements are 2 bytes (bf16). Tensor dims set generously (no OOB clipping).
__device__ __forceinline__ void tdm_load_2d(const bf16_t* lds_dst,
                                            const bf16_t* gsrc,
                                            unsigned tile0, unsigned tile1,
                                            unsigned long long stride0) {
    unsigned lds_off = (unsigned)(uintptr_t)lds_dst;   // flat LDS addr low 32b = offset
    unsigned long long ga = (unsigned long long)(uintptr_t)gsrc;

    u32x4 g0;
    g0[0] = 1u;                                         // count=1, user desc
    g0[1] = lds_off;                                    // lds_addr (bytes)
    g0[2] = (unsigned)ga;                               // global_addr[31:0]
    g0[3] = ((unsigned)(ga >> 32) & 0x1FFFFFFu)         // global_addr[56:32]
            | (2u << 30);                               // type = 2 ("image")

    const unsigned long long td0 = 1ull << 20;          // tensor_dim0 (generous)
    const unsigned long long td1 = 1ull << 20;          // tensor_dim1 (generous)
    i32x8 g1;
    g1[0] = (int)(1u << 16);                            // data_size=1 (2 bytes)
    g1[1] = (int)(((unsigned)td0 & 0xFFFFu) << 16);     // tensor_dim0[15:0]
    g1[2] = (int)(((unsigned)(td0 >> 16) & 0xFFFFu) |
                  (((unsigned)td1 & 0xFFFFu) << 16));   // td0[31:16] | td1[15:0]
    g1[3] = (int)((((unsigned)(td1 >> 16)) & 0xFFFFu) |
                  (tile0 << 16));                       // td1[31:16] | tile_dim0
    g1[4] = (int)(tile1 & 0xFFFFu);                     // tile_dim1 (tile_dim2=0)
    g1[5] = (int)(unsigned)stride0;                     // dim0_stride[31:0]
    g1[6] = (int)((unsigned)(stride0 >> 32) & 0xFFFFu); // dim0_stride[47:32]
    g1[7] = 0;

    i32x4 z4 = {0, 0, 0, 0};
#if defined(__clang_major__) && (__clang_major__ >= 23)
    i32x8 z8 = {0, 0, 0, 0, 0, 0, 0, 0};
    __builtin_amdgcn_tensor_load_to_lds(g0, g1, z4, z4, z8, 0);
#else
    __builtin_amdgcn_tensor_load_to_lds(g0, g1, z4, z4, 0);
#endif
}

// ---------------------------------------------------------------- converts
__global__ __launch_bounds__(256) void cvt_x_kernel(const float* __restrict__ x,
                                                    bf16_t* __restrict__ xb, int n) {
    int i = blockIdx.x * 256 + threadIdx.x;
    if (i < n) xb[i] = (bf16_t)x[i];
}

// w_q/k/v: [h][d][e] f32 -> transposed bf16 [h][e][d]
__global__ __launch_bounds__(256) void cvt_wqkv_kernel(const float* __restrict__ wq,
                                                       const float* __restrict__ wk,
                                                       const float* __restrict__ wv,
                                                       bf16_t* __restrict__ wqt,
                                                       bf16_t* __restrict__ wkt,
                                                       bf16_t* __restrict__ wvt) {
    int i = blockIdx.x * 256 + threadIdx.x;        // over NHEAD*D_MODEL*DQKV
    int h   = i >> 16;                             // D_MODEL*DQKV == 65536
    int rem = i & 65535;
    int d   = rem >> 6;
    int e   = rem & 63;
    size_t dst = ((size_t)h * DQKV + e) * D_MODEL + d;
    wqt[dst] = (bf16_t)wq[i];
    wkt[dst] = (bf16_t)wk[i];
    wvt[dst] = (bf16_t)wv[i];
}

// w_o: [k][n] f32 -> transposed bf16 [n][k]
__global__ __launch_bounds__(256) void cvt_wo_kernel(const float* __restrict__ wo,
                                                     bf16_t* __restrict__ wot) {
    int i = blockIdx.x * 256 + threadIdx.x;        // over D_MODEL*D_MODEL
    int k = i >> 10;
    int n = i & 1023;
    wot[(size_t)n * D_MODEL + k] = (bf16_t)wo[i];
}

// ---------------------------------------------------------------- QKV projection
// block = one 16-row x panel (TDM-staged in LDS) x 8 (head,mat) combos
__global__ __launch_bounds__(256) void qkv_proj_kernel(const bf16_t* __restrict__ xb,
                                                       const bf16_t* __restrict__ wqt,
                                                       const bf16_t* __restrict__ wkt,
                                                       const bf16_t* __restrict__ wvt,
                                                       bf16_t* __restrict__ qb,
                                                       bf16_t* __restrict__ kb,
                                                       bf16_t* __restrict__ vt) {
    __shared__ __align__(128) bf16_t sx[16 * D_MODEL];  // 32 KB A panel

    const int wv   = threadIdx.x >> 5;
    const int lane = threadIdx.x & 31;
    const int half = lane >> 4;
    const int ln   = lane & 15;

    const int MT   = MTOT / 16;                    // 256
    int mt    = blockIdx.x % MT;
    int grp   = blockIdx.x / MT;                   // 0..5
    int combo = grp * 8 + wv;                      // 0..47
    int h     = combo & 15;
    int mat   = combo >> 4;                        // 0=q 1=k 2=v

    if (threadIdx.x < 32) {
        tdm_load_2d(sx, xb + (size_t)mt * 16 * D_MODEL, D_MODEL, 16, D_MODEL);
        __builtin_amdgcn_s_wait_tensorcnt(0);
    }
    __syncthreads();

    const bf16_t* wt = (mat == 0 ? wqt : (mat == 1 ? wkt : wvt)) +
                       (size_t)h * DQKV * D_MODEL;

    v8f acc[4];
#pragma unroll
    for (int t = 0; t < 4; ++t) acc[t] = vzero8();

    for (int k0 = 0; k0 < D_MODEL; k0 += 32) {
        const bf16_t* xr = &sx[ln * D_MODEL + k0];
        v16bf a = cat16(ld8(xr + half * 16), ld8(xr + half * 16 + 8));
#pragma unroll
        for (int t = 0; t < 4; ++t) {
            v16bf bw = *(const v16bf*)(wt + (size_t)(t * 16 + ln) * D_MODEL +
                                       k0 + half * 16);
            acc[t] = wmma_bf16(a, bw, acc[t]);
        }
    }

#pragma unroll
    for (int r = 0; r < 8; ++r) {
        int m = mt * 16 + half * 8 + r;
        int b = m >> 11;                           // /SEQ
        int l = m & (SEQ - 1);
#pragma unroll
        for (int t = 0; t < 4; ++t) {
            int e = t * 16 + ln;
            float v = acc[t][r];
            if (mat == 0) {                        // fold 1/sqrt(64) into Q
                qb[((size_t)((b * NHEAD + h) * SEQ) + l) * DQKV + e] =
                    (bf16_t)(v * 0.125f);
            } else if (mat == 1) {
                kb[((size_t)((b * NHEAD + h) * SEQ) + l) * DQKV + e] = (bf16_t)v;
            } else {                               // V transposed [b][h][e][l]
                vt[((size_t)((b * NHEAD + h) * DQKV) + e) * SEQ + l] = (bf16_t)v;
            }
        }
    }
}

// ---------------------------------------------------------------- flash attention
// 8 waves/block share (b,h); K/V chunks double-buffered in LDS via TDM.
__global__ __launch_bounds__(256) void attn_fa_kernel(const bf16_t* __restrict__ qb,
                                                      const bf16_t* __restrict__ kb,
                                                      const bf16_t* __restrict__ vt,
                                                      const unsigned char* __restrict__ mask,
                                                      bf16_t* __restrict__ ctx) {
    __shared__ __align__(128) bf16_t skbuf[2][32 * DQKV];   // 2 x 4 KB
    __shared__ __align__(128) bf16_t svbuf[2][DQKV * 32];   // 2 x 4 KB
    __shared__ __align__(128) bf16_t pstage[8][16 * 32];    // 8 KB

    const int wv   = threadIdx.x >> 5;
    const int lane = threadIdx.x & 31;
    const int half = lane >> 4;
    const int ln   = lane & 15;

    const int QT = SEQ / 16;                       // 128 (8 | 128 => block shares b,h)
    int g  = blockIdx.x * 8 + wv;
    int qt = g % QT;
    int h  = (g / QT) % NHEAD;
    int b  = g / (QT * NHEAD);

    const bf16_t* kbase = kb + (size_t)((b * NHEAD + h) * SEQ) * DQKV;
    const bf16_t* vbase = vt + (size_t)((b * NHEAD + h) * DQKV) * SEQ;

    // Q A-operands (two 16x32 slabs), loaded once
    const bf16_t* qrow = qb + ((size_t)((b * NHEAD + h) * SEQ) + qt * 16 + ln) * DQKV;
    v16bf aq0 = cat16(ld8(qrow + half * 16),      ld8(qrow + half * 16 + 8));
    v16bf aq1 = cat16(ld8(qrow + 32 + half * 16), ld8(qrow + 40 + half * 16));

    float rmax[8], rsum[8];
    v8f   acc[4];
#pragma unroll
    for (int r = 0; r < 8; ++r) { rmax[r] = -1e30f; rsum[r] = 0.0f; }
#pragma unroll
    for (int t = 0; t < 4; ++t) acc[t] = vzero8();

    const unsigned char* mrow =
        mask + ((size_t)b * SEQ + qt * 16 + half * 8) * SEQ;

    // prime buffer 0
    if (threadIdx.x < 32) {
        tdm_load_2d(&skbuf[0][0], kbase, 32 * DQKV, 0, 32 * DQKV); // 1D 32x64
        tdm_load_2d(&svbuf[0][0], vbase, 32, DQKV, SEQ);           // 64 rows x 32
    }

    for (int jc = 0; jc < SEQ; jc += 32) {
        const int cur = (jc >> 5) & 1;
        if (threadIdx.x < 32) {
            if (jc + 32 < SEQ) {        // prefetch next chunk, keep 1 in flight
                tdm_load_2d(&skbuf[cur ^ 1][0], kbase + (size_t)(jc + 32) * DQKV,
                            32 * DQKV, 0, 32 * DQKV);
                tdm_load_2d(&svbuf[cur ^ 1][0], vbase + (jc + 32), 32, DQKV, SEQ);
                __builtin_amdgcn_s_wait_tensorcnt(2);  // current chunk done
            } else {
                __builtin_amdgcn_s_wait_tensorcnt(0);
            }
        }
        __syncthreads();

        const bf16_t* ks = &skbuf[cur][0];
        const bf16_t* vs = &svbuf[cur][0];

        // ---- logits for 32 keys: two 16x16 C tiles, K-dim = e (64)
        v8f c0 = vzero8(), c1 = vzero8();
        c0 = wmma_bf16(aq0, *(const v16bf*)(ks + (size_t)ln * DQKV + half * 16), c0);
        c0 = wmma_bf16(aq1, *(const v16bf*)(ks + (size_t)ln * DQKV + 32 + half * 16), c0);
        c1 = wmma_bf16(aq0, *(const v16bf*)(ks + (size_t)(16 + ln) * DQKV + half * 16), c1);
        c1 = wmma_bf16(aq1, *(const v16bf*)(ks + (size_t)(16 + ln) * DQKV + 32 + half * 16), c1);

        // ---- mask (True => -1e9)
#pragma unroll
        for (int r = 0; r < 8; ++r) {
            if (mrow[(size_t)r * SEQ + jc + ln])      c0[r] = -1e9f;
            if (mrow[(size_t)r * SEQ + jc + 16 + ln]) c1[r] = -1e9f;
        }

        // ---- online softmax (row stats live in 16-lane halves)
#pragma unroll
        for (int r = 0; r < 8; ++r) {
            float tm = fmaxf(c0[r], c1[r]);
#pragma unroll
            for (int m = 1; m < 16; m <<= 1) tm = fmaxf(tm, __shfl_xor(tm, m, 16));
            float nm   = fmaxf(rmax[r], tm);
            float corr = __expf(rmax[r] - nm);
            rmax[r] = nm;
            float p0 = __expf(c0[r] - nm);
            float p1 = __expf(c1[r] - nm);
            float ps = p0 + p1;
#pragma unroll
            for (int m = 1; m < 16; m <<= 1) ps += __shfl_xor(ps, m, 16);
            rsum[r] = rsum[r] * corr + ps;
#pragma unroll
            for (int t = 0; t < 4; ++t) acc[t][r] *= corr;
            pstage[wv][(half * 8 + r) * 32 + ln]      = (bf16_t)p0;
            pstage[wv][(half * 8 + r) * 32 + 16 + ln] = (bf16_t)p1;
        }

        // C-layout -> A-layout through LDS (same-wave DS ordering)
        v16bf ap = cat16(*(const v8bf*)&pstage[wv][ln * 32 + half * 16],
                         *(const v8bf*)&pstage[wv][ln * 32 + half * 16 + 8]);

        // ---- context += P(16x32) * V(32x16), 4 e-tiles
#pragma unroll
        for (int t = 0; t < 4; ++t) {
            v16bf bv = *(const v16bf*)(vs + (size_t)(t * 16 + ln) * 32 + half * 16);
            acc[t] = wmma_bf16(ap, bv, acc[t]);
        }
        __syncthreads();   // buffers reusable
    }

    // ---- epilogue: normalize, emit ctx[b*L+l][h*64+e] bf16
#pragma unroll
    for (int r = 0; r < 8; ++r) {
        float inv = 1.0f / rsum[r];
        size_t row = (size_t)(b * SEQ + qt * 16 + half * 8 + r);
#pragma unroll
        for (int t = 0; t < 4; ++t)
            ctx[row * D_MODEL + h * DQKV + t * 16 + ln] =
                (bf16_t)(acc[t][r] * inv);
    }
}

// ---------------------------------------------------------------- out-proj + residual
// block = one 16-row ctx panel (TDM-staged); wave wv covers N-tiles [wv*8, wv*8+8)
__global__ __launch_bounds__(256) void out_proj_kernel(const bf16_t* __restrict__ ctx,
                                                       const bf16_t* __restrict__ wot,
                                                       const float* __restrict__ x,
                                                       float* __restrict__ y) {
    __shared__ __align__(128) bf16_t sc[16 * D_MODEL];  // 32 KB A panel

    const int wv   = threadIdx.x >> 5;
    const int lane = threadIdx.x & 31;
    const int half = lane >> 4;
    const int ln   = lane & 15;
    const int mt   = blockIdx.x;

    if (threadIdx.x < 32) {
        tdm_load_2d(sc, ctx + (size_t)mt * 16 * D_MODEL, D_MODEL, 16, D_MODEL);
        __builtin_amdgcn_s_wait_tensorcnt(0);
    }
    __syncthreads();

    v8f acc[8];
#pragma unroll
    for (int t = 0; t < 8; ++t) acc[t] = vzero8();

    for (int k0 = 0; k0 < D_MODEL; k0 += 32) {
        const bf16_t* cr = &sc[ln * D_MODEL + k0];
        v16bf a = cat16(ld8(cr + half * 16), ld8(cr + half * 16 + 8));
#pragma unroll
        for (int t = 0; t < 8; ++t) {
            int n0 = (wv * 8 + t) * 16;
            v16bf bw = *(const v16bf*)(wot + (size_t)(n0 + ln) * D_MODEL +
                                       k0 + half * 16);
            acc[t] = wmma_bf16(a, bw, acc[t]);
        }
    }

#pragma unroll
    for (int r = 0; r < 8; ++r) {
        size_t row = (size_t)(mt * 16 + half * 8 + r);
#pragma unroll
        for (int t = 0; t < 8; ++t) {
            int col = (wv * 8 + t) * 16 + ln;
            y[row * D_MODEL + col] = acc[t][r] + x[row * D_MODEL + col];
        }
    }
}

// ---------------------------------------------------------------- layernorm
__global__ __launch_bounds__(256) void ln_kernel(const float* __restrict__ y,
                                                 const float* __restrict__ gamma,
                                                 const float* __restrict__ beta,
                                                 float* __restrict__ out) {
    __shared__ float s1[256];
    __shared__ float s2[256];
    const int row = blockIdx.x;
    const int tid = threadIdx.x;
    const float* yr = y + (size_t)row * D_MODEL;

    float a = 0.0f, b = 0.0f;
    for (int c = tid; c < D_MODEL; c += 256) { float v = yr[c]; a += v; b += v * v; }
    s1[tid] = a; s2[tid] = b;
    __syncthreads();
    for (int s = 128; s > 0; s >>= 1) {
        if (tid < s) { s1[tid] += s1[tid + s]; s2[tid] += s2[tid + s]; }
        __syncthreads();
    }
    const float mu   = s1[0] * (1.0f / D_MODEL);
    const float var  = s2[0] * (1.0f / D_MODEL) - mu * mu;
    const float rstd = rsqrtf(var + LN_EPS);
    for (int c = tid; c < D_MODEL; c += 256)
        out[(size_t)row * D_MODEL + c] = (yr[c] - mu) * rstd * gamma[c] + beta[c];
}

// ---------------------------------------------------------------- launch
extern "C" void kernel_launch(void* const* d_in, const int* in_sizes, int n_in,
                              void* d_out, int out_size, void* d_ws, size_t ws_size,
                              hipStream_t stream) {
    const float*         x     = (const float*)d_in[0];
    const unsigned char* mask  = (const unsigned char*)d_in[1];
    const float*         wq    = (const float*)d_in[2];
    const float*         wk    = (const float*)d_in[3];
    const float*         wvw   = (const float*)d_in[4];
    const float*         wo    = (const float*)d_in[5];
    const float*         gamma = (const float*)d_in[6];
    const float*         beta  = (const float*)d_in[7];

    char* p = (char*)d_ws;
    bf16_t* xb  = (bf16_t*)p; p += (size_t)MTOT * D_MODEL * 2;          //  8 MiB
    bf16_t* wqt = (bf16_t*)p; p += (size_t)NHEAD * DQKV * D_MODEL * 2;  //  2 MiB
    bf16_t* wkt = (bf16_t*)p; p += (size_t)NHEAD * DQKV * D_MODEL * 2;
    bf16_t* wvt = (bf16_t*)p; p += (size_t)NHEAD * DQKV * D_MODEL * 2;
    bf16_t* wot = (bf16_t*)p; p += (size_t)D_MODEL * D_MODEL * 2;       //  2 MiB
    bf16_t* qb  = (bf16_t*)p; p += (size_t)MTOT * DQKV * NHEAD * 2;     //  8 MiB
    bf16_t* kb  = (bf16_t*)p; p += (size_t)MTOT * DQKV * NHEAD * 2;
    bf16_t* vt  = (bf16_t*)p; p += (size_t)MTOT * DQKV * NHEAD * 2;
    bf16_t* ctx = (bf16_t*)p; p += (size_t)MTOT * D_MODEL * 2;          //  8 MiB
    float*  y   = (float*)p;                                            // 16 MiB

    cvt_x_kernel<<<(MTOT * D_MODEL + 255) / 256, 256, 0, stream>>>(
        x, xb, MTOT * D_MODEL);
    cvt_wqkv_kernel<<<(NHEAD * D_MODEL * DQKV) / 256, 256, 0, stream>>>(
        wq, wk, wvw, wqt, wkt, wvt);
    cvt_wo_kernel<<<(D_MODEL * D_MODEL) / 256, 256, 0, stream>>>(wo, wot);

    // 256 row tiles x 6 wave-groups (48 (h,mat) combos / 8 waves)
    qkv_proj_kernel<<<(MTOT / 16) * 6, 256, 0, stream>>>(
        xb, wqt, wkt, wvt, qb, kb, vt);

    // 2*16*128 = 4096 q-tile waves / 8 per block
    attn_fa_kernel<<<512, 256, 0, stream>>>(qb, kb, vt, mask, ctx);

    out_proj_kernel<<<MTOT / 16, 256, 0, stream>>>(ctx, wot, x, y);

    ln_kernel<<<MTOT, 256, 0, stream>>>(y, gamma, beta, (float*)d_out);
}
